// EncoderDecoder_1194000908408
// MI455X (gfx1250) — matrix operand order repaired and checked by reading, not compiled, not run on hardware
//
#include <hip/hip_runtime.h>
#include <hip/hip_bf16.h>

typedef __attribute__((ext_vector_type(16))) _Float16 v16h;
typedef __attribute__((ext_vector_type(8)))  _Float16 v8h;
typedef __attribute__((ext_vector_type(8)))  float    v8f;

#define DEVI __device__ __forceinline__

DEVI float sigm(float x) { return 1.0f / (1.0f + __expf(-x)); }

// gfx1250 async global->LDS copy (16B per lane) + wait, via inline asm
// (portable across ROCm7.2 / amdgpu-toolchain; builtin arity differs).
DEVI void async_ld_b128(unsigned ldsoff, const void* gptr) {
  asm volatile("global_load_async_to_lds_b128 %0, %1, off"
               :: "v"(ldsoff), "v"(gptr) : "memory");
}
DEVI void wait_async0() { asm volatile("s_wait_asynccnt 0x0" ::: "memory"); }

// ---------------------------------------------------------------------------
// WMMA GEMM:  out[n*ldc + m] = act( sum_k W[m][k]*X[n][k] + bias[m] )
//   W : f16 row-major [Mp][K], Mp = gridDim.x*128 (zero-padded rows if M<Mp)
//   X : f32 or f16 row-major [N][K], N = gridDim.y*16, K multiple of 32 (<=1120)
// Block = 256 threads = 8 waves. The 16-row X panel is staged once in LDS
// (async global->LDS when X is f16), then each wave computes a 16x16 tile
// with v_wmma_f32_16x16x32_f16, B-fragments read via ds_load_b128.
// Fragment maps (CDNA5 ISA 7.12.2, wave32):
//   A 16x32 f16 : lane(l) M=l%16, g=l/16; elt e<8 -> K=8g+e ; e>=8 -> 16+8g+(e-8)
//   B 32x16 f16 : lane(l) N=l%16, g=l/16; elt e<8 -> K=8g+e ; e>=8 -> 16+8g+(e-8)
//   C 16x16 f32 : lane(l) N=l%16; vgpr r -> M=8*(l/16)+r
// ---------------------------------------------------------------------------
template <typename XT, int ACT> // ACT: 0=none 1=relu 2=tanh
__global__ __launch_bounds__(256) void gemm_wmma(
    const _Float16* __restrict__ Wp, const XT* __restrict__ Xm,
    const float* __restrict__ bias, float* __restrict__ Cm,
    int M, int K, long ldc) {
  extern __shared__ _Float16 smem[];           // [16][K+8]
  const int tid = threadIdx.x;
  const int lane = tid & 31, wave = tid >> 5;
  const int g = lane >> 4, lm = lane & 15;
  const int n0 = blockIdx.y * 16;
  const int Kp = K + 8;

  // ---- stage X[n0..n0+15][0..K) into LDS ----
  if constexpr (sizeof(XT) == 2) {
    // f16: async DMA, 16B chunks; row r keeps +16B pad in LDS
    unsigned sbase = (unsigned)(size_t)(void*)smem;   // low 32b = LDS offset
    int cpr = K >> 3;                                  // 16B chunks per row
    for (int tch = tid; tch < 16 * cpr; tch += 256) {
      int r = tch / cpr, ch = tch - r * cpr;
      unsigned ldsoff = sbase + (unsigned)(r * Kp + ch * 8) * 2u;
      const void* gp = (const void*)(Xm + (size_t)(n0 + r) * K + ch * 8);
      async_ld_b128(ldsoff, gp);
    }
    wait_async0();
  } else {
    for (int i = tid; i < 16 * K; i += 256) {
      int r = i / K, k = i - r * K;
      smem[r * Kp + k] = (_Float16)Xm[(size_t)(n0 + r) * K + k];
    }
  }
  __syncthreads();

  // ---- per-wave 16x16 tile ----
  const int m0 = blockIdx.x * 128 + wave * 16;
  v8f acc = {};
  const _Float16* wr = Wp + (size_t)(m0 + lm) * K + g * 8;
  const _Float16* br = smem + lm * Kp + g * 8;
  for (int k0 = 0; k0 < K; k0 += 32) {
    __builtin_prefetch(wr + k0 + 64, 0, 1);
    v8h alo = *(const v8h*)(wr + k0);
    v8h ahi = *(const v8h*)(wr + k0 + 16);
    v8h blo = *(const v8h*)(br + k0);      // ds_load_b128
    v8h bhi = *(const v8h*)(br + k0 + 16); // ds_load_b128
    v16h a, b;
#pragma unroll
    for (int e = 0; e < 8; ++e) {
      a[e] = alo[e]; a[e + 8] = ahi[e];
      b[e] = blo[e]; b[e + 8] = bhi[e];
    }
    acc = __builtin_amdgcn_wmma_f32_16x16x32_f16(false, a, false, b,
                                                 (short)0, acc, false, false);
  }
#pragma unroll
  for (int r = 0; r < 8; ++r) {
    int m = m0 + g * 8 + r;
    int n = n0 + lm;
    float v = acc[r];
    if (bias) v += bias[m];
    if (ACT == 1) v = v > 0.f ? v : 0.f;
    if (ACT == 2) v = tanhf(v);
    if (m < M) Cm[(size_t)n * ldc + m] = v;
  }
}

// ---------------------------------------------------------------------------
// Implicit-GEMM 3x3 SAME conv (NHWC, f16 in/out, f32 accum) + bias + ReLU.
// Wp: packed f16 weights [Cout][K], K = 9*Cin, k = (kh*3+kw)*Cin + ci.
// Block = 256 threads = 8 waves; grid = (Cout/128, W/16, B*H).
// The gathered im2col panel (16 x-positions, Kc=576 chunk) is staged in LDS
// cooperatively, then 8 waves each run a 16(Cout)x16(x) WMMA tile.
// ---------------------------------------------------------------------------
#define CONV_KC  576
#define CONV_KCP (CONV_KC + 8)
__global__ __launch_bounds__(256) void conv_wmma(
    const _Float16* __restrict__ Wp, const _Float16* __restrict__ Fin,
    const float* __restrict__ bias, _Float16* __restrict__ Fout,
    int H, int W, int Cin, int log2Cin, int Cout) {
  __shared__ _Float16 sb[16 * CONV_KCP];
  const int tid = threadIdx.x;
  const int lane = tid & 31, wave = tid >> 5;
  const int g = lane >> 4, lm = lane & 15;
  const int co0 = blockIdx.x * 128 + wave * 16;
  const int x0 = blockIdx.y * 16;
  const int b = blockIdx.z / H, y = blockIdx.z % H;
  const int K = 9 << log2Cin;
  v8f acc = {};
  const _Float16* wr = Wp + (size_t)(co0 + lm) * K + g * 8;

  for (int kc = 0; kc < K; kc += CONV_KC) {
    // ---- cooperative im2col gather into LDS ----
    for (int i = tid; i < 16 * CONV_KC; i += 256) {
      int r = i / CONV_KC, kk = i - r * CONV_KC;
      int k = kc + kk;
      int ci = k & (Cin - 1);
      int khkw = k >> log2Cin;        // 0..8
      int kh = khkw / 3, kw = khkw - kh * 3;
      int yy = y + kh - 1, xx = x0 + r + kw - 1;
      _Float16 v = (_Float16)0.f;
      if (yy >= 0 && yy < H && xx >= 0 && xx < W)
        v = Fin[((((size_t)b * H + yy) * W + xx) << log2Cin) + ci];
      sb[r * CONV_KCP + kk] = v;
    }
    __syncthreads();
    // ---- WMMA over this K chunk ----
    const _Float16* br = sb + lm * CONV_KCP + g * 8;
    const _Float16* wc = wr + kc;
    for (int k0 = 0; k0 < CONV_KC; k0 += 32) {
      __builtin_prefetch(wc + k0 + 64, 0, 1);
      v8h alo = *(const v8h*)(wc + k0);
      v8h ahi = *(const v8h*)(wc + k0 + 16);
      v8h blo = *(const v8h*)(br + k0);
      v8h bhi = *(const v8h*)(br + k0 + 16);
      v16h a, bb;
#pragma unroll
      for (int e = 0; e < 8; ++e) {
        a[e] = alo[e]; a[e + 8] = ahi[e];
        bb[e] = blo[e]; bb[e + 8] = bhi[e];
      }
      acc = __builtin_amdgcn_wmma_f32_16x16x32_f16(false, a, false, bb,
                                                   (short)0, acc, false, false);
    }
    __syncthreads();
  }
  size_t sp = (((size_t)b * H + y) * W + (x0 + lm));
#pragma unroll
  for (int r = 0; r < 8; ++r) {
    int co = co0 + g * 8 + r;
    float v = acc[r] + bias[co];
    v = v > 0.f ? v : 0.f;
    Fout[sp * Cout + co] = (_Float16)v;
  }
}

// ------------------------- encoder helper kernels --------------------------
__global__ void conv1_pool(const float* __restrict__ X, const float* __restrict__ w1,
                           const float* __restrict__ b1, _Float16* __restrict__ out) {
  int idx = blockIdx.x * 256 + threadIdx.x;
  const int total = 32 * 32 * 256 * 64;
  if (idx >= total) return;
  int co = idx & 63;
  int x2 = (idx >> 6) & 255;
  int y2 = (idx >> 14) & 31;
  int b = idx >> 19;
  float wv[9];
#pragma unroll
  for (int i = 0; i < 9; ++i) wv[i] = w1[co * 9 + i];
  float bs = b1[co];
  float m = -1e30f;
#pragma unroll
  for (int dy = 0; dy < 2; ++dy)
#pragma unroll
    for (int dx = 0; dx < 2; ++dx) {
      int y = y2 * 2 + dy, x = x2 * 2 + dx;
      float s = bs;
#pragma unroll
      for (int kh = 0; kh < 3; ++kh)
#pragma unroll
        for (int kw = 0; kw < 3; ++kw) {
          int yy = y + kh - 1, xx = x + kw - 1;
          if (yy >= 0 && yy < 64 && xx >= 0 && xx < 512)
            s += X[((size_t)b * 64 + yy) * 512 + xx] * wv[kh * 3 + kw];
        }
      s = s > 0.f ? s : 0.f;
      m = fmaxf(m, s);
    }
  out[idx] = (_Float16)m;
}

__global__ void pool2_f16(const _Float16* __restrict__ in, _Float16* __restrict__ out,
                          int B, int H, int W, int C) {
  int Ho = H >> 1, Wo = W >> 1;
  int total = B * Ho * Wo * C;
  int idx = blockIdx.x * 256 + threadIdx.x;
  if (idx >= total) return;
  int cc = idx % C;
  int t = idx / C;
  int x2 = t % Wo; t /= Wo;
  int y2 = t % Ho; int b = t / Ho;
  const _Float16* base = in + ((((size_t)b * H + y2 * 2) * W) + x2 * 2) * C + cc;
  float v0 = (float)base[0], v1 = (float)base[C];
  float v2 = (float)base[(size_t)W * C], v3 = (float)base[(size_t)W * C + C];
  out[idx] = (_Float16)fmaxf(fmaxf(v0, v1), fmaxf(v2, v3));
}

// ------------------------- weight packing kernels --------------------------
__global__ void pack_convw(const float* __restrict__ w, _Float16* __restrict__ dst,
                           int Cout, int Cin) {
  int K = Cin * 9;
  int total = Cout * K;
  int idx = blockIdx.x * 256 + threadIdx.x;
  if (idx >= total) return;
  int co = idx / K, k = idx - co * K;
  int khkw = k / Cin, ci = k - khkw * Cin;
  dst[idx] = (_Float16)w[((size_t)co * Cin + ci) * 9 + khkw];
}

__global__ void transpose_cvt(const float* __restrict__ src, _Float16* __restrict__ dst,
                              int rows, int cols) {
  int idx = blockIdx.x * 256 + threadIdx.x;
  if (idx >= rows * cols) return;
  int r = idx / cols, cc = idx - r * cols;
  dst[(size_t)cc * rows + r] = (_Float16)src[idx];
}

__global__ void cvt_f16(const float* __restrict__ src, _Float16* __restrict__ dst, int n) {
  int idx = blockIdx.x * 256 + threadIdx.x;
  if (idx < n) dst[idx] = (_Float16)src[idx];
}

__global__ void pack_wcat(const float* __restrict__ Wx, const float* __restrict__ Wh,
                          _Float16* __restrict__ dst) {
  int idx = blockIdx.x * 256 + threadIdx.x;
  if (idx >= 2048 * 1120) return;
  int m = idx / 1120, k = idx - m * 1120;
  float v = 0.f;
  if (k < 592) v = Wx[(size_t)m * 592 + k];
  else if (k < 1104) v = Wh[(size_t)m * 512 + (k - 592)];
  dst[idx] = (_Float16)v;
}

__global__ void pack_wout(const float* __restrict__ W_out, _Float16* __restrict__ dst) {
  int idx = blockIdx.x * 256 + threadIdx.x;
  if (idx >= 512 * 512) return;
  int m = idx >> 9, k = idx & 511;
  dst[idx] = (m < 500) ? (_Float16)W_out[(size_t)m * 512 + k] : (_Float16)0.f;
}

__global__ void zero_f32(float* p, int n) {
  int idx = blockIdx.x * 256 + threadIdx.x;
  if (idx < n) p[idx] = 0.f;
}

// ------------------------- decoder pointwise kernels -----------------------
__global__ void concat_xh(const int* __restrict__ labels, const float* __restrict__ E,
                          const float* __restrict__ O, const float* __restrict__ h,
                          float* __restrict__ xh, int t) {
  int idx = blockIdx.x * 256 + threadIdx.x;
  if (idx >= 32 * 1120) return;
  int b = idx / 1120, k = idx - b * 1120;
  float v = 0.f;
  if (k < 80) {
    int id = (t == 0) ? 497 : labels[b * 150 + (t - 1)];
    v = E[id * 80 + k];
  } else if (k < 592) v = O[b * 512 + (k - 80)];
  else if (k < 1104)  v = h[b * 512 + (k - 592)];
  xh[idx] = v;
}

__global__ void lstm_pw(const float* __restrict__ gates, float* __restrict__ c,
                        float* __restrict__ h) {
  int idx = blockIdx.x * 256 + threadIdx.x;
  if (idx >= 32 * 512) return;
  int b = idx >> 9, j = idx & 511;
  const float* gb = gates + (size_t)b * 2048;
  float ig = sigm(gb[j]), fg = sigm(gb[512 + j]);
  float gg = tanhf(gb[1024 + j]), og = sigm(gb[1536 + j]);
  float cn = fg * c[idx] + ig * gg;
  c[idx] = cn;
  h[idx] = og * tanhf(cn);
}

__global__ __launch_bounds__(256) void att_score_softmax(
    const float* __restrict__ Vproj, const float* __restrict__ hW,
    const float* __restrict__ beta, float* __restrict__ attA) {
  int b = blockIdx.x, tid = threadIdx.x;
  __shared__ float hwS[512], beS[512], sc[512], red[256];
  for (int j = tid; j < 512; j += 256) { hwS[j] = hW[b * 512 + j]; beS[j] = beta[j]; }
  __syncthreads();
  for (int l = tid; l < 512; l += 256) {
    const float* vp = Vproj + ((size_t)b * 512 + l) * 512;
    float s = 0.f;
    for (int j = 0; j < 512; ++j) s += tanhf(vp[j] + hwS[j]) * beS[j];
    sc[l] = s;
  }
  __syncthreads();
  red[tid] = fmaxf(sc[tid], sc[tid + 256]);
  __syncthreads();
  for (int s2 = 128; s2 > 0; s2 >>= 1) {
    if (tid < s2) red[tid] = fmaxf(red[tid], red[tid + s2]);
    __syncthreads();
  }
  float mx = red[0];
  __syncthreads();
  float e0 = __expf(sc[tid] - mx), e1 = __expf(sc[tid + 256] - mx);
  red[tid] = e0 + e1;
  __syncthreads();
  for (int s2 = 128; s2 > 0; s2 >>= 1) {
    if (tid < s2) red[tid] += red[tid + s2];
    __syncthreads();
  }
  float inv = 1.f / red[0];
  attA[b * 512 + tid] = e0 * inv;
  attA[b * 512 + tid + 256] = e1 * inv;
}

__global__ __launch_bounds__(256) void ctx_xo(
    const float* __restrict__ attA, const _Float16* __restrict__ V,
    const float* __restrict__ h, float* __restrict__ xo) {
  int b = blockIdx.x, tid = threadIdx.x;
  __shared__ float aS[512];
  for (int l = tid; l < 512; l += 256) aS[l] = attA[b * 512 + l];
  __syncthreads();
  for (int cc = tid; cc < 512; cc += 256) {
    const _Float16* vb = V + ((size_t)b * 512) * 512 + cc;
    float s = 0.f;
    for (int l = 0; l < 512; ++l) s += aS[l] * (float)vb[(size_t)l * 512];
    xo[b * 1024 + 512 + cc] = s;
    xo[b * 1024 + cc] = h[b * 512 + cc];
  }
}

// ---------------------------------------------------------------------------
extern "C" void kernel_launch(void* const* d_in, const int* in_sizes, int n_in,
                              void* d_out, int out_size, void* d_ws, size_t ws_size,
                              hipStream_t stream) {
  (void)in_sizes; (void)n_in; (void)out_size; (void)ws_size;
  const float* X      = (const float*)d_in[0];
  const int*   labels = (const int*)d_in[1];
  const float* w1 = (const float*)d_in[2];
  const float* b1 = (const float*)d_in[3];
  const float* w2 = (const float*)d_in[4];
  const float* b2 = (const float*)d_in[5];
  const float* w3 = (const float*)d_in[6];
  const float* b3 = (const float*)d_in[7];
  const float* w4 = (const float*)d_in[8];
  const float* b4 = (const float*)d_in[9];
  const float* E      = (const float*)d_in[10];
  const float* Wx     = (const float*)d_in[11];
  const float* Wh     = (const float*)d_in[12];
  const float* b_lstm = (const float*)d_in[13];
  const float* Wv     = (const float*)d_in[14];
  const float* Wha    = (const float*)d_in[15];
  const float* b_att  = (const float*)d_in[16];
  const float* beta   = (const float*)d_in[17];
  const float* W_O    = (const float*)d_in[18];
  const float* W_out  = (const float*)d_in[19];
  float* out = (float*)d_out;

  char* ws = (char*)d_ws;
  size_t off = 0;
  auto alloc = [&](size_t bytes) -> void* {
    void* p = ws + off;
    off += (bytes + 255) & ~(size_t)255;
    return p;
  };
  _Float16* F1p   = (_Float16*)alloc(16777216ull * 2); // (32,32,256,64) / conv3 out
  _Float16* BIG   = (_Float16*)alloc(33554432ull * 2); // conv2 raw / conv4 raw
  _Float16* F2p   = (_Float16*)alloc(8388608ull * 2);  // (32,16,128,128)
  _Float16* Vb    = (_Float16*)alloc(8388608ull * 2);  // V: (32, 512, 512)
  float*    Vproj = (float*)alloc(8388608ull * 4);     // (32*512, 512)
  _Float16* wp2   = (_Float16*)alloc(73728ull * 2);
  _Float16* wp3   = (_Float16*)alloc(294912ull * 2);
  _Float16* wp4   = (_Float16*)alloc(1179648ull * 2);
  _Float16* Wcat  = (_Float16*)alloc(2293760ull * 2);  // [2048][1120]
  _Float16* WvT   = (_Float16*)alloc(262144ull * 2);
  _Float16* WhT   = (_Float16*)alloc(262144ull * 2);
  _Float16* WOh   = (_Float16*)alloc(524288ull * 2);
  _Float16* WoutP = (_Float16*)alloc(262144ull * 2);
  float* xh    = (float*)alloc(32 * 1120 * 4);
  float* gates = (float*)alloc(32 * 2048 * 4);
  float* hbuf  = (float*)alloc(32 * 512 * 4);
  float* cbuf  = (float*)alloc(32 * 512 * 4);
  float* Obuf  = (float*)alloc(32 * 512 * 4);
  float* hW    = (float*)alloc(32 * 512 * 4);
  float* attA  = (float*)alloc(32 * 512 * 4);
  float* xo    = (float*)alloc(32 * 1024 * 4);

  auto blks = [](int n) { return (n + 255) / 256; };
  auto smem16 = [](int K) { return (size_t)(16 * (K + 8) * 2); };

  // ---- weight packing / state init ----
  pack_convw<<<blks(128 * 576), 256, 0, stream>>>(w2, wp2, 128, 64);
  pack_convw<<<blks(256 * 1152), 256, 0, stream>>>(w3, wp3, 256, 128);
  pack_convw<<<blks(512 * 2304), 256, 0, stream>>>(w4, wp4, 512, 256);
  pack_wcat<<<blks(2048 * 1120), 256, 0, stream>>>(Wx, Wh, Wcat);
  transpose_cvt<<<blks(512 * 512), 256, 0, stream>>>(Wv, WvT, 512, 512);
  transpose_cvt<<<blks(512 * 512), 256, 0, stream>>>(Wha, WhT, 512, 512);
  cvt_f16<<<blks(512 * 1024), 256, 0, stream>>>(W_O, WOh, 512 * 1024);
  pack_wout<<<blks(512 * 512), 256, 0, stream>>>(W_out, WoutP);
  zero_f32<<<blks(32 * 512), 256, 0, stream>>>(hbuf, 32 * 512);
  zero_f32<<<blks(32 * 512), 256, 0, stream>>>(cbuf, 32 * 512);
  zero_f32<<<blks(32 * 512), 256, 0, stream>>>(Obuf, 32 * 512);

  // ---- CNN encoder ----
  conv1_pool<<<blks(32 * 32 * 256 * 64), 256, 0, stream>>>(X, w1, b1, F1p);
  conv_wmma<<<dim3(1, 16, 32 * 32), 256, 0, stream>>>(wp2, F1p, b2, BIG,
                                                      32, 256, 64, 6, 128);
  pool2_f16<<<blks(32 * 16 * 128 * 128), 256, 0, stream>>>(BIG, F2p, 32, 32, 256, 128);
  _Float16* C3 = F1p; // reuse
  conv_wmma<<<dim3(2, 8, 32 * 16), 256, 0, stream>>>(wp3, F2p, b3, C3,
                                                     16, 128, 128, 7, 256);
  conv_wmma<<<dim3(4, 8, 32 * 16), 256, 0, stream>>>(wp4, C3, b4, BIG,
                                                     16, 128, 256, 8, 512);
  pool2_f16<<<blks(32 * 8 * 64 * 512), 256, 0, stream>>>(BIG, Vb, 32, 16, 128, 512);

  // Vproj = V @ Wv_att + b_att   (M=512, K=512, N=16384; f16 X -> async staging)
  gemm_wmma<_Float16, 0><<<dim3(4, 1024), 256, smem16(512), stream>>>(
      WvT, Vb, b_att, Vproj, 512, 512, 512);

  // ---- decoder: 150 sequential steps ----
  for (int t = 0; t < 150; ++t) {
    concat_xh<<<blks(32 * 1120), 256, 0, stream>>>(labels, E, Obuf, hbuf, xh, t);
    gemm_wmma<float, 0><<<dim3(16, 2), 256, smem16(1120), stream>>>(
        Wcat, xh, b_lstm, gates, 2048, 1120, 2048);
    lstm_pw<<<blks(32 * 512), 256, 0, stream>>>(gates, cbuf, hbuf);
    gemm_wmma<float, 0><<<dim3(4, 2), 256, smem16(512), stream>>>(
        WhT, hbuf, nullptr, hW, 512, 512, 512);
    att_score_softmax<<<32, 256, 0, stream>>>(Vproj, hW, beta, attA);
    ctx_xo<<<32, 256, 0, stream>>>(attA, Vb, hbuf, xo);
    gemm_wmma<float, 2><<<dim3(4, 2), 256, smem16(1024), stream>>>(
        WOh, xo, nullptr, Obuf, 512, 1024, 512);
    gemm_wmma<float, 0><<<dim3(4, 2), 256, smem16(512), stream>>>(
        WoutP, Obuf, nullptr, out + (size_t)t * 500, 500, 512, 150L * 500L);
  }
}
